// MultiheadAttention_12627203850808
// MI455X (gfx1250) — matrix operand order repaired and checked by reading, not compiled, not run on hardware
//
#include <hip/hip_runtime.h>

// ---------------------------------------------------------------------------
// MHA forward for MI455X (gfx1250), wave32 + WMMA bf16 (fp32 accumulate).
//   d_out = [ O : B*S*D  |  attention : B*H*S*S ]  (fp32)
// Pipeline:
//   1) Qp = query @ Wq^T ; Kp = key @ Wk^T ; Vp = value @ Wv^T   (TN gemm)
//   2) energy = (Qp_h @ Kp_h^T) / 8   -> d_out attn region       (batched TN)
//   3) masked softmax in-place on attn region
//   4) ctx_h = attn_h @ Vp_h          (batched, B is [K,N])
//   5) O = ctx @ Wo^T + bo                                        (TN gemm)
// Double-buffered LDS software pipeline, v_perm_b32 bf16 packing.
// ---------------------------------------------------------------------------

typedef __attribute__((ext_vector_type(16))) __bf16 v16bf;
typedef __attribute__((ext_vector_type(8)))  float  v8f;

union FragBF {
    v16bf v;
    uint4 u[2];
};

#define BM 128
#define BK 32
#define KPAD 40   // 32 + 8 pad; row = 80B, keeps 16B alignment, dodges bank conflicts

// Pack two fp32 -> packed bf16 pair: one v_perm_b32 after round-half-up bias.
__device__ __forceinline__ unsigned int pkbf(float x, float y) {
    return __builtin_amdgcn_perm(__float_as_uint(y) + 0x8000u,
                                 __float_as_uint(x) + 0x8000u,
                                 0x07060302u);
}
__device__ __forceinline__ unsigned short f2bf(float f) {
    return (unsigned short)((__float_as_uint(f) + 0x8000u) >> 16);
}
__device__ __forceinline__ void store4bf(unsigned short* d, float4 f) {
    *(uint2*)d = make_uint2(pkbf(f.x, f.y), pkbf(f.z, f.w));
}

// C[m][n] = alpha * sum_k A[m][k] * B(n,k) + bias[n]
//   TRANS_B = true : B stored [N][K] row-major (weights, K^T for energy)
//   TRANS_B = false: B stored [K][N] row-major (V for attn@V); transposed in LDS
// Batched over blockIdx.z: z -> (zb=z/HH, zh=z%HH), offsets zb*s?b + zh*s?h.
template <int BN_, bool TRANS_B>
__global__ void __launch_bounds__(256)
gemm_bf16_wmma(const float* __restrict__ A, const float* __restrict__ Bm,
               float* __restrict__ C, const float* __restrict__ bias,
               int K, int lda, int ldb, int ldc, int HH,
               long long sAb, long long sAh,
               long long sBb, long long sBh,
               long long sCb, long long sCh,
               float alpha) {
    constexpr int NT    = BN_ / 32;                 // wmma col-tiles per wave
    constexpr int BITER = TRANS_B ? (BN_ / 32) : (BN_ * BK / 4 / 256);

    const int z  = blockIdx.z;
    const int zb = z / HH;
    const int zh = z % HH;
    A  += (long long)zb * sAb + (long long)zh * sAh;
    Bm += (long long)zb * sBb + (long long)zh * sBh;
    C  += (long long)zb * sCb + (long long)zh * sCh;

    const int m0 = blockIdx.y * BM;
    const int n0 = blockIdx.x * BN_;

    const int tid  = threadIdx.x;
    const int lane = tid & 31;
    const int wave = tid >> 5;
    const int wm   = wave & 3;   // 4 waves along M (each 32 rows)
    const int wn   = wave >> 2;  // 2 waves along N (each BN_/2 cols)

    __shared__ unsigned short As[2][BM][KPAD];
    __shared__ unsigned short Bs[2][BN_][KPAD];

    const v8f vzero = {0.f, 0.f, 0.f, 0.f, 0.f, 0.f, 0.f, 0.f};
    v8f acc[2][NT];
    #pragma unroll
    for (int mt = 0; mt < 2; ++mt)
        #pragma unroll
        for (int nt = 0; nt < NT; ++nt) acc[mt][nt] = vzero;

    const int half = (lane < 16) ? 0 : 1;
    const int lrow = lane & 15;
    const int nK   = K / BK;

    float4 ra[4];
    float4 rb[BITER];

    auto loadG = [&](int kt) {
        const int k0 = kt * BK;
        #pragma unroll
        for (int it = 0; it < 4; ++it) {            // A: 128x32 = 1024 float4
            int idx = tid + it * 256;
            int r = idx >> 3, c4 = (idx & 7) * 4;
            ra[it] = *(const float4*)(A + (long long)(m0 + r) * lda + k0 + c4);
        }
        if constexpr (TRANS_B) {
            #pragma unroll
            for (int it = 0; it < BITER; ++it) {    // B[n][k]
                int idx = tid + it * 256;
                int r = idx >> 3, c4 = (idx & 7) * 4;
                rb[it] = *(const float4*)(Bm + (long long)(n0 + r) * ldb + k0 + c4);
            }
        } else {
            #pragma unroll
            for (int it = 0; it < BITER; ++it) {    // B[k][n]
                int idx = tid + it * 256;
                int kk = idx >> 4, n4 = (idx & 15) * 4;
                rb[it] = *(const float4*)(Bm + (long long)(k0 + kk) * ldb + n0 + n4);
            }
        }
    };

    auto storeS = [&](int buf) {
        #pragma unroll
        for (int it = 0; it < 4; ++it) {
            int idx = tid + it * 256;
            int r = idx >> 3, c4 = (idx & 7) * 4;
            store4bf(&As[buf][r][c4], ra[it]);
        }
        if constexpr (TRANS_B) {
            #pragma unroll
            for (int it = 0; it < BITER; ++it) {
                int idx = tid + it * 256;
                int r = idx >> 3, c4 = (idx & 7) * 4;
                store4bf(&Bs[buf][r][c4], rb[it]);
            }
        } else {
            #pragma unroll
            for (int it = 0; it < BITER; ++it) {    // transpose [k][n] -> [n][k]
                int idx = tid + it * 256;
                int kk = idx >> 4, n4 = (idx & 15) * 4;
                Bs[buf][n4 + 0][kk] = f2bf(rb[it].x);
                Bs[buf][n4 + 1][kk] = f2bf(rb[it].y);
                Bs[buf][n4 + 2][kk] = f2bf(rb[it].z);
                Bs[buf][n4 + 3][kk] = f2bf(rb[it].w);
            }
        }
    };

    auto compute = [&](int buf) {
        // A 16x32 frag: lane<16 -> K {0..7}+{16..23}; lane>=16 -> {8..15}+{24..31}
        FragBF a[2], b[NT];
        #pragma unroll
        for (int mt = 0; mt < 2; ++mt) {
            const unsigned short* p = &As[buf][wm * 32 + mt * 16 + lrow][half * 8];
            a[mt].u[0] = *(const uint4*)(p);
            a[mt].u[1] = *(const uint4*)(p + 16);
        }
        // B 32x16 frag: lane<16 -> K {0..7}+{16..23} of column lrow; lane>=16 -> {8..15}+{24..31}
        #pragma unroll
        for (int nt = 0; nt < NT; ++nt) {
            const unsigned short* p = &Bs[buf][wn * (BN_ / 2) + nt * 16 + lrow][half * 8];
            b[nt].u[0] = *(const uint4*)(p);
            b[nt].u[1] = *(const uint4*)(p + 16);
        }
        #pragma unroll
        for (int mt = 0; mt < 2; ++mt)
            #pragma unroll
            for (int nt = 0; nt < NT; ++nt)
                acc[mt][nt] = __builtin_amdgcn_wmma_f32_16x16x32_bf16(
                    false, a[mt].v, false, b[nt].v,
                    (short)0, acc[mt][nt], false, false);
    };

    // software pipeline: global loads of tile k+1 overlap WMMAs on tile k
    loadG(0);
    storeS(0);
    __syncthreads();
    for (int kt = 0; kt < nK; ++kt) {
        const int cur = kt & 1;
        const bool more = (kt + 1 < nK);
        if (more) loadG(kt + 1);
        compute(cur);
        if (more) storeS(cur ^ 1);
        __syncthreads();
    }

    // ---- epilogue: C/D 16x16 layout -> lane<16 rows g, lane>=16 rows g+8
    const int roff = (lane < 16) ? 0 : 8;
    #pragma unroll
    for (int mt = 0; mt < 2; ++mt) {
        #pragma unroll
        for (int nt = 0; nt < NT; ++nt) {
            const int col  = n0 + wn * (BN_ / 2) + nt * 16 + lrow;
            const int row0 = m0 + wm * 32 + mt * 16 + roff;
            const float bv = bias ? bias[col] : 0.0f;
            #pragma unroll
            for (int g = 0; g < 8; ++g)
                C[(long long)(row0 + g) * ldc + col] = acc[mt][nt][g] * alpha + bv;
        }
    }
}

// In-place masked softmax over rows of attn [B*H*S, S], mask [B,S,S] int32.
__global__ void __launch_bounds__(256)
softmax_mask_kernel(float* __restrict__ attn, const int* __restrict__ mask) {
    const int S = 2048, H = 16;
    const long long r = blockIdx.x;          // row id in [0, B*H*S)
    const int q = (int)(r % S);
    const int b = (int)(r / ((long long)H * S));
    float*     row  = attn + r * (long long)S;
    const int* mrow = mask + ((long long)b * S + q) * (long long)S;
    const int tid = threadIdx.x;

    float v[8];
    float lmax = -3.4e38f;
    #pragma unroll
    for (int j = 0; j < 8; ++j) {
        int i = tid + j * 256;
        float e = row[i];
        v[j] = (mrow[i] == 0) ? -1e30f : e;
        lmax = fmaxf(lmax, v[j]);
    }
    __shared__ float red[256];
    red[tid] = lmax;
    __syncthreads();
    for (int s = 128; s > 0; s >>= 1) {
        if (tid < s) red[tid] = fmaxf(red[tid], red[tid + s]);
        __syncthreads();
    }
    const float m = red[0];
    __syncthreads();
    float lsum = 0.f;
    #pragma unroll
    for (int j = 0; j < 8; ++j) {
        v[j] = __expf(v[j] - m);
        lsum += v[j];
    }
    red[tid] = lsum;
    __syncthreads();
    for (int s = 128; s > 0; s >>= 1) {
        if (tid < s) red[tid] += red[tid + s];
        __syncthreads();
    }
    const float inv = 1.0f / red[0];
    #pragma unroll
    for (int j = 0; j < 8; ++j)
        row[tid + j * 256] = v[j] * inv;
}

extern "C" void kernel_launch(void* const* d_in, const int* in_sizes, int n_in,
                              void* d_out, int out_size, void* d_ws, size_t ws_size,
                              hipStream_t stream) {
    (void)in_sizes; (void)n_in; (void)out_size; (void)ws_size;
    constexpr int       Bd = 2, S = 2048, D = 1024, H = 16, DK = 64;
    constexpr long long ROWS = (long long)Bd * S;          // 4096
    constexpr long long MATN = ROWS * D;                   // 4,194,304

    const float* query = (const float*)d_in[0];
    const float* key   = (const float*)d_in[1];
    const float* value = (const float*)d_in[2];
    const int*   mask  = (const int*)d_in[3];
    const float* Wq = (const float*)d_in[4];
    const float* Wk = (const float*)d_in[5];
    const float* Wv = (const float*)d_in[6];
    const float* Wo = (const float*)d_in[7];
    const float* bo = (const float*)d_in[8];

    float* O    = (float*)d_out;
    float* attn = O + MATN;                    // [B,H,S,S]
    float* Qp   = (float*)d_ws;                // 4 x 16MB fp32 scratch
    float* Kp   = Qp + MATN;
    float* Vp   = Kp + MATN;
    float* ctx  = Vp + MATN;

    // 1) projections: M=4096, N=1024, K=1024, C = X @ W^T (128x128 tiles)
    dim3 gproj(D / 128, (int)ROWS / BM, 1);
    gemm_bf16_wmma<128, true><<<gproj, 256, 0, stream>>>(
        query, Wq, Qp, nullptr, D, D, D, D, 1, 0, 0, 0, 0, 0, 0, 1.0f);
    gemm_bf16_wmma<128, true><<<gproj, 256, 0, stream>>>(
        key, Wk, Kp, nullptr, D, D, D, D, 1, 0, 0, 0, 0, 0, 0, 1.0f);
    gemm_bf16_wmma<128, true><<<gproj, 256, 0, stream>>>(
        value, Wv, Vp, nullptr, D, D, D, D, 1, 0, 0, 0, 0, 0, 0, 1.0f);

    // 2) energy = Q_h @ K_h^T / sqrt(dk): per (b,h); M=N=2048, K=64
    dim3 genergy(S / 128, S / BM, Bd * H);
    gemm_bf16_wmma<128, true><<<genergy, 256, 0, stream>>>(
        Qp, Kp, attn, nullptr, DK, D, D, S, H,
        (long long)S * D, DK,
        (long long)S * D, DK,
        (long long)H * S * S, (long long)S * S,
        0.125f);

    // 3) masked softmax in place (one block per row)
    softmax_mask_kernel<<<Bd * H * S, 256, 0, stream>>>(attn, mask);

    // 4) ctx_h = attn_h @ V_h : M=2048, N=64, K=2048; B stored [K,N]
    dim3 gav(DK / 64, S / BM, Bd * H);
    gemm_bf16_wmma<64, false><<<gav, 256, 0, stream>>>(
        attn, Vp, ctx, nullptr, S, S, D, D, H,
        (long long)H * S * S, (long long)S * S,
        (long long)S * D, DK,
        (long long)S * D, DK,
        1.0f);

    // 5) O = ctx @ Wo^T + bo : M=4096, N=1024, K=1024
    gemm_bf16_wmma<128, true><<<gproj, 256, 0, stream>>>(
        ctx, Wo, O, bo, D, D, D, D, 1, 0, 0, 0, 0, 0, 0, 1.0f);
}
